// GNN_model_83837761617950
// MI455X (gfx1250) — matrix operand order repaired and checked by reading, not compiled, not run on hardware
//
#include <hip/hip_runtime.h>
#include <math.h>

typedef __attribute__((ext_vector_type(16))) _Float16 v16h;
typedef __attribute__((ext_vector_type(8)))  float    v8f;

#define DIM 128
#define DIM3 384

// ---------------------------------------------------------------- utilities

__global__ void fill_f32_kernel(float* __restrict__ p, float v, int n) {
    int t = blockIdx.x * blockDim.x + threadIdx.x;
    if (t < n) p[t] = v;
}

__global__ void f32_to_f16_kernel(const float* __restrict__ src,
                                  _Float16* __restrict__ dst, int n) {
    int t = blockIdx.x * blockDim.x + threadIdx.x;
    if (t < n) dst[t] = (_Float16)src[t];
}

// W is [K=128][N=128] row-major; produce Wt[n][k] in f16 so that the WMMA
// B-fragment loads (per-column, K-contiguous) are coalesced 32-bit loads.
__global__ void w_transpose_f16_kernel(const float* __restrict__ W,
                                       _Float16* __restrict__ Wt) {
    int t = blockIdx.x * blockDim.x + threadIdx.x;   // 128*128 threads
    if (t >= DIM * DIM) return;
    int k = t / DIM, n = t % DIM;
    Wt[n * DIM + k] = (_Float16)W[k * DIM + n];
}

// ---------------------------------------------------------------- GEMM (WMMA)
// H[N x 128] = Xh[N x 128] * W  (W supplied transposed as Wt[n][k], f16).
// Block = 8 waves, covers a 32-row x 128-col output tile.
//   - A tile (32x128 f16 = 8 KB) staged once in LDS (kills the 8x redundant
//     per-wave global A loads; fragment reads become ds_load_b128).
//   - Each wave owns one 16-col tile and produces TWO 16x16 row tiles,
//     reusing its B fragments in registers: 8 v_wmma per wave.
__global__ __launch_bounds__(256)
void gemm_wmma_kernel(const _Float16* __restrict__ Xh,
                      const _Float16* __restrict__ Wt,
                      float* __restrict__ H, int nrows) {
    __shared__ _Float16 As[32 * DIM];          // 8 KB

    const int tid  = threadIdx.x;
    const int wave = tid >> 5;                 // 0..7 -> column tile
    const int lane = tid & 31;
    const int half = lane >> 4;                // 0: lanes 0-15, 1: lanes 16-31
    const int l16  = lane & 15;
    const int row0 = blockIdx.x * 32;

    // Cooperative A-tile copy: 32 rows x 128 halfs; 16 halfs (32 B) per thread.
    {
        int r = tid >> 3;                      // 0..31 (8 threads per row)
        int c = (tid & 7) * 16;                // 16-half chunk
        int gr = row0 + r;
        if (gr >= nrows) gr = nrows - 1;       // clamp (tail block)
        const float4* s4 = (const float4*)(Xh + (size_t)gr * DIM + c);
        float4* d4 = (float4*)(As + r * DIM + c);
        d4[0] = s4[0];
        d4[1] = s4[1];
    }
    __syncthreads();

    const int col = wave * 16 + l16;
    const _Float16* __restrict__ bx = Wt + (size_t)col * DIM;
    const _Float16* a0 = As + l16 * DIM;       // rows 0-15 of tile
    const _Float16* a1 = As + (l16 + 16) * DIM;// rows 16-31 of tile

    v8f acc0 = {}, acc1 = {};
    #pragma unroll
    for (int kk = 0; kk < DIM; kk += 32) {
        v16h a, a2, b;
        #pragma unroll
        for (int v = 0; v < 8; ++v) {
            // A fragment: ISA 16-bit A 16x32 layout.
            //   lanes 0-15 : V0..3 -> K=2v,2v+1 ; V4..7 -> K=16+2(v-4)..
            //   lanes16-31 : same + 8
            int ka = kk + half * 8 + ((v < 4) ? (2 * v) : (16 + 2 * (v - 4)));
            a [2 * v] = a0[ka];  a [2 * v + 1] = a0[ka + 1];
            a2[2 * v] = a1[ka];  a2[2 * v + 1] = a1[ka + 1];
            // B fragment: same K layout as A (V0..3 -> K=2v,2v+1 ; V4..7 -> K=16+2(v-4).., lanes16-31 +8).
            int kb = kk + half * 8 + ((v < 4) ? (2 * v) : (16 + 2 * (v - 4)));
            b[2 * v] = bx[kb];   b[2 * v + 1] = bx[kb + 1];
        }
        acc0 = __builtin_amdgcn_wmma_f32_16x16x32_f16(
                   false, a,  false, b, (short)0, acc0, false, false);
        acc1 = __builtin_amdgcn_wmma_f32_16x16x32_f16(
                   false, a2, false, b, (short)0, acc1, false, false);
    }

    // C/D layout: lanes 0-15 VGPR r -> M=r ; lanes 16-31 -> M=r+8 ; N=lane&15.
    if (row0 + 32 <= nrows) {
        // fast path: branch-free stores, one base pointer, immediate offsets
        float* out0 = H + (size_t)(row0 + 8 * half) * DIM + col;
        #pragma unroll
        for (int r = 0; r < 8; ++r) out0[r * DIM] = acc0[r];
        float* out1 = out0 + 16 * DIM;
        #pragma unroll
        for (int r = 0; r < 8; ++r) out1[r * DIM] = acc1[r];
    } else {
        #pragma unroll
        for (int r = 0; r < 8; ++r) {
            int orow = row0 + r + 8 * half;
            if (orow < nrows)      H[(size_t)orow * DIM + col]        = acc0[r];
            if (orow + 16 < nrows) H[(size_t)(orow + 16) * DIM + col] = acc1[r];
        }
    }
}

// ---------------------------------------------------------------- graph ops

__global__ void degree_kernel(const int* __restrict__ dst, float* __restrict__ deg,
                              int nedges) {
    int e = blockIdx.x * blockDim.x + threadIdx.x;
    if (e < nedges) atomicAdd(&deg[dst[e]], 1.0f);
}

__global__ void rsqrt_kernel(const float* __restrict__ deg, float* __restrict__ dinv,
                             int n) {
    int t = blockIdx.x * blockDim.x + threadIdx.x;
    if (t < n) dinv[t] = rsqrtf(deg[t]);
}

// One edge per 32-lane group; lane L handles features [4L, 4L+4).
__global__ __launch_bounds__(256)
void edge_scatter_kernel(const float* __restrict__ h, const int* __restrict__ src,
                         const int* __restrict__ dst, const float* __restrict__ dinv,
                         float* __restrict__ agg, int nedges) {
    long long t = (long long)blockIdx.x * blockDim.x + threadIdx.x;
    int e = (int)(t >> 5);
    if (e >= nedges) return;
    int lane = (int)(t & 31);
    int s = src[e], d = dst[e];
    float nrm = dinv[s] * dinv[d];
    const float4 v = ((const float4*)(h + (size_t)s * DIM))[lane];
    float* ap = agg + (size_t)d * DIM + lane * 4;
    atomicAdd(ap + 0, v.x * nrm);
    atomicAdd(ap + 1, v.y * nrm);
    atomicAdd(ap + 2, v.z * nrm);
    atomicAdd(ap + 3, v.w * nrm);
}

// out = relu(agg + h * dinv^2 + bias); write f32 in place and f16 for next GEMM.
__global__ void combine_relu_kernel(float* __restrict__ h, const float* __restrict__ agg,
                                    const float* __restrict__ dinv,
                                    const float* __restrict__ bias,
                                    _Float16* __restrict__ xh_next, int nnodes) {
    long long t = (long long)blockIdx.x * blockDim.x + threadIdx.x;
    if (t >= (long long)nnodes * DIM) return;
    int i = (int)(t / DIM), j = (int)(t % DIM);
    float di = dinv[i];
    float v = agg[t] + h[t] * di * di + bias[j];
    v = fmaxf(v, 0.0f);
    h[t] = v;
    xh_next[t] = (_Float16)v;
}

__global__ void pool_sum_kernel(const float* __restrict__ h1, const float* __restrict__ h2,
                                const float* __restrict__ h3, const int* __restrict__ batch,
                                float* __restrict__ pooled, float* __restrict__ cnt,
                                int nnodes) {
    long long t = (long long)blockIdx.x * blockDim.x + threadIdx.x;
    if (t >= (long long)nnodes * DIM3) return;
    int i = (int)(t / DIM3), j = (int)(t % DIM3);
    const float* hb = (j < DIM) ? h1 : ((j < 2 * DIM) ? h2 : h3);
    float v = hb[(size_t)i * DIM + (j & (DIM - 1))];
    int g = batch[i];
    atomicAdd(&pooled[(size_t)g * DIM3 + j], v);
    if (j == 0) atomicAdd(&cnt[g], 1.0f);
}

__global__ void finalize_kernel(const float* __restrict__ p1, const float* __restrict__ c1,
                                const float* __restrict__ p2, const float* __restrict__ c2,
                                float* __restrict__ out, int ngraphs) {
    int g = blockIdx.x * blockDim.x + threadIdx.x;
    if (g >= ngraphs) return;
    float i1 = 1.0f / fmaxf(c1[g], 1.0f);
    float i2 = 1.0f / fmaxf(c2[g], 1.0f);
    float s = 0.0f;
    for (int j = 0; j < DIM3; ++j) {
        float d = p1[(size_t)g * DIM3 + j] * i1 - p2[(size_t)g * DIM3 + j] * i2;
        s += d * d;
    }
    out[g] = sqrtf(s);
}

// ---------------------------------------------------------------- launch

static inline int cdiv(long long a, int b) { return (int)((a + b - 1) / b); }

extern "C" void kernel_launch(void* const* d_in, const int* in_sizes, int n_in,
                              void* d_out, int out_size, void* d_ws, size_t ws_size,
                              hipStream_t stream) {
    const float* x[2]  = { (const float*)d_in[0], (const float*)d_in[1] };
    const float* W[3]  = { (const float*)d_in[2], (const float*)d_in[4], (const float*)d_in[6] };
    const float* B[3]  = { (const float*)d_in[3], (const float*)d_in[5], (const float*)d_in[7] };
    const int*   ei[2] = { (const int*)d_in[8], (const int*)d_in[9] };
    const int*   bv[2] = { (const int*)d_in[10], (const int*)d_in[11] };

    const int N = in_sizes[0] / DIM;      // 50000
    const int E = in_sizes[8] / 2;        // 600000
    const int G = out_size;               // 64

    // scratch carve-out
    char* w = (char*)d_ws;
    size_t off = 0;
    auto take = [&](size_t bytes) -> char* {
        char* p = w + off;
        off = (off + bytes + 255) & ~(size_t)255;
        return p;
    };
    _Float16* xh   = (_Float16*)take((size_t)N * DIM * 2);
    _Float16* wht  = (_Float16*)take((size_t)DIM * DIM * 2);
    float* hbuf[3];
    hbuf[0] = (float*)take((size_t)N * DIM * 4);
    hbuf[1] = (float*)take((size_t)N * DIM * 4);
    hbuf[2] = (float*)take((size_t)N * DIM * 4);
    float* agg    = (float*)take((size_t)N * DIM * 4);
    float* deg    = (float*)take((size_t)N * 4);
    float* dinv   = (float*)take((size_t)N * 4);
    float* pooled[2] = { (float*)take((size_t)G * DIM3 * 4), (float*)take((size_t)G * DIM3 * 4) };
    float* cnt[2]    = { (float*)take((size_t)G * 4),        (float*)take((size_t)G * 4) };
    (void)ws_size; (void)n_in;

    const int BT = 256;

    for (int p = 0; p < 2; ++p) {
        const int* src = ei[p];
        const int* dst = ei[p] + E;

        // degrees -> dinv (deg starts at 1 for the self loop)
        fill_f32_kernel<<<cdiv(N, BT), BT, 0, stream>>>(deg, 1.0f, N);
        degree_kernel<<<cdiv(E, BT), BT, 0, stream>>>(dst, deg, E);
        rsqrt_kernel<<<cdiv(N, BT), BT, 0, stream>>>(deg, dinv, N);

        // pooled accumulators
        fill_f32_kernel<<<cdiv(G * DIM3, BT), BT, 0, stream>>>(pooled[p], 0.0f, G * DIM3);
        fill_f32_kernel<<<cdiv(G, BT), BT, 0, stream>>>(cnt[p], 0.0f, G);

        // layer input: x in f16
        f32_to_f16_kernel<<<cdiv((long long)N * DIM, BT), BT, 0, stream>>>(x[p], xh, N * DIM);

        for (int l = 0; l < 3; ++l) {
            w_transpose_f16_kernel<<<cdiv(DIM * DIM, BT), BT, 0, stream>>>(W[l], wht);
            gemm_wmma_kernel<<<cdiv(N, 32), BT, 0, stream>>>(xh, wht, hbuf[l], N);
            fill_f32_kernel<<<cdiv((long long)N * DIM, BT), BT, 0, stream>>>(agg, 0.0f, N * DIM);
            edge_scatter_kernel<<<cdiv((long long)E * 32, BT), BT, 0, stream>>>(
                hbuf[l], src, dst, dinv, agg, E);
            combine_relu_kernel<<<cdiv((long long)N * DIM, BT), BT, 0, stream>>>(
                hbuf[l], agg, dinv, B[l], xh, N);
        }

        pool_sum_kernel<<<cdiv((long long)N * DIM3, BT), BT, 0, stream>>>(
            hbuf[0], hbuf[1], hbuf[2], bv[p], pooled[p], cnt[p], N);
    }

    finalize_kernel<<<cdiv(G, 64), 64, 0, stream>>>(
        pooled[0], cnt[0], pooled[1], cnt[1], (float*)d_out, G);
}
